// HybridGCN_86397562126412
// MI455X (gfx1250) — matrix operand order repaired and checked by reading, compiled, not run on hardware
//
#include <hip/hip_runtime.h>
#include <math.h>
#include <stdint.h>

// ---------------- problem constants (match reference) ----------------
#define NN    8192
#define EE    262144
#define DINC  256
#define HIDC  512
#define ELOOP (EE + NN)        // edges incl. self loops = 270336
#define NM    (NN * DINC)      // 2097152 elems per 8192x256 matrix

// ---------------- WMMA / TDM types (gfx1250, wave32) ----------------
typedef __attribute__((ext_vector_type(16))) __bf16        v16bf;
typedef __attribute__((ext_vector_type(8)))  float         v8f;
typedef __attribute__((ext_vector_type(4)))  unsigned int  u32x4;
typedef __attribute__((ext_vector_type(8)))  unsigned int  u32x8;

union Frag16 { v16bf v; unsigned short u[16]; };

__device__ __forceinline__ unsigned short f2bf(float f) {
  // float -> bf16, round-to-nearest-even (finite inputs)
  unsigned int u = __float_as_uint(f);
  u += 0x7fffu + ((u >> 16) & 1u);
  return (unsigned short)(u >> 16);
}

__device__ __forceinline__ void atomicMaxF(float* addr, float val) {
  if (val >= 0.0f) atomicMax((int*)addr, __float_as_int(val));
  else             atomicMin((unsigned int*)addr, (unsigned int)__float_as_int(val));
}

// =====================================================================
// TDM: DMA one bf16 tile (128 rows x 32 cols, row stride `ld` elems) from
// global memory into LDS at byte offset lds_off, with LDS rows padded
// 64B data + 16B pad (matches LDS row stride of 40 bf16 = 80 B).
// Descriptor per CDNA5 ISA:
//   group0: count=1 | lds_addr | global_addr(57b) | type=2
//   group1: data_size=2B, pad_enable, pad_interval=16 DW, pad_amount=4 DW,
//           tensor_dim=(32,128) == tile_dim (tile always fully in-bounds),
//           tensor_dim0_stride = ld
// Issued wave-uniform (wave 0 only); tracked by TENSORcnt.
// =====================================================================
__device__ __forceinline__ void tdm_load_tile(const unsigned short* gbase,
                                              unsigned int ld, unsigned int lds_off)
{
  unsigned long long ga = (unsigned long long)(uintptr_t)gbase;
  u32x4 g0;
  g0[0] = 1u;                                   // count=1, user descriptor
  g0[1] = lds_off;                              // LDS byte address
  g0[2] = (unsigned int)ga;                     // global_addr[31:0]
  g0[3] = (unsigned int)((ga >> 32) & 0x01ffffffu) | (2u << 30);  // addr[56:32] | type=2
  u32x8 g1;
  g1[0] = (1u << 16) | (1u << 20) | (3u << 22) | (3u << 25); // ds=2B, pad, ivl=16DW, amt=4DW
  g1[1] = 32u << 16;                            // tensor_dim0 = 32   (bits 79:48 lo)
  g1[2] = 128u << 16;                           // tensor_dim1 = 128  (bits 111:80 lo)
  g1[3] = 32u << 16;                            // tile_dim0 = 32     (bits 127:112)
  g1[4] = 128u;                                 // tile_dim1 = 128, tile_dim2 = 0
  g1[5] = ld;                                   // tensor_dim0_stride[31:0]
  g1[6] = 0u;                                   // stride hi / dim1_stride lo
  g1[7] = 0u;
  asm volatile("tensor_load_to_lds %0, %1" :: "s"(g0), "s"(g1) : "memory");
}

// =====================================================================
// Tiled bf16 WMMA GEMM with TDM-staged, double-buffered LDS tiles.
//   C[M x N] = act( A[M x K] @ B^T + bias ),  BT is pre-transposed [N][K] bf16
//   A split across two row-major bf16 sources: col k<K1 -> A1 else A2
//   Block = 256 threads (8 waves). Tile 128x128, K-step 32.
//   Wave w computes rows [w*16,w*16+16) x 128 = 8 WMMA tiles.
//   All dims exact multiples (M%128==N%128==0, K%32==0, TK | K1).
// =====================================================================
#define TK 32
#define LDS_S 40        // LDS row stride in bf16 (80 B: 16B-aligned, conflict-free)
#define TILE_US (128 * LDS_S)   // 5120 ushorts = 10240 B per tile buffer

__global__ __launch_bounds__(256)
void hybridgcn_gemm(const unsigned short* __restrict__ A1, int lda1,
                    const unsigned short* __restrict__ A2, int lda2, int K1,
                    const unsigned short* __restrict__ BT,
                    const float* __restrict__ bias,
                    void* __restrict__ C, int ldc,
                    int K, int relu, int bf16out)
{
  __shared__ unsigned short smem[4 * TILE_US];   // [buf0:A,B][buf1:A,B] = 40 KB

  const int t     = threadIdx.x;
  const int wave  = t >> 5;
  const int lane  = t & 31;
  const int lhalf = lane >> 4;
  const int l15   = lane & 15;
  const int m0    = blockIdx.y * 128;
  const int n0    = blockIdx.x * 128;

  const unsigned int lds_base = (unsigned int)(uintptr_t)(void*)smem;

  v8f acc[8] = {};

  const int arow = wave * 16 + l15;   // A-frag row in tile
  const int akb  = lhalf * 8;         // A-frag K base (ISA 7.12.2, 16-bit A 16x32)
  const int bkb  = lhalf * 8;         // B-frag K base

  const int nsteps = K / TK;

  // ---- prologue: DMA tile 0 into buffer 0 ----
  if (wave == 0) {
    const unsigned short* a0 = (0 < K1) ? A1 + (size_t)m0 * lda1
                                        : A2 + (size_t)m0 * lda2;
    int la0 = (0 < K1) ? lda1 : lda2;
    tdm_load_tile(a0, (unsigned)la0, lds_base + 0 * 2 * (TILE_US * 2));
    tdm_load_tile(BT + (size_t)n0 * K, (unsigned)K,
                  lds_base + (0 * 2 + 1) * (TILE_US * 2));
    __builtin_amdgcn_s_wait_tensorcnt((short)0);
  }
  __syncthreads();

  for (int s = 0; s < nsteps; ++s) {
    const int cur = s & 1;
    // ---- prefetch next tile into the other buffer (wave 0, TDM) ----
    if (wave == 0 && s + 1 < nsteps) {
      int k0 = (s + 1) * TK;
      const unsigned short* an; int lan;
      if (k0 < K1) { an = A1 + (size_t)m0 * lda1 + k0;        lan = lda1; }
      else         { an = A2 + (size_t)m0 * lda2 + (k0 - K1); lan = lda2; }
      int nb = cur ^ 1;
      tdm_load_tile(an, (unsigned)lan, lds_base + (nb * 2)     * (TILE_US * 2));
      tdm_load_tile(BT + (size_t)n0 * K + k0, (unsigned)K,
                    lds_base + (nb * 2 + 1) * (TILE_US * 2));
    }

    const unsigned short* Asc = smem + (cur * 2)     * TILE_US;
    const unsigned short* Bsc = smem + (cur * 2 + 1) * TILE_US;

    // ---- A fragment: lane holds M=arow, K = akb+0..7 and 16+akb+0..7 ----
    Frag16 a;
    #pragma unroll
    for (int j = 0; j < 8; j++) {
      int k = (j < 4) ? (akb + 2 * j) : (16 + akb + 2 * j - 8);
      unsigned int pr = *(const unsigned int*)&Asc[arow * LDS_S + k];
      a.u[2 * j]     = (unsigned short)(pr & 0xffffu);
      a.u[2 * j + 1] = (unsigned short)(pr >> 16);
    }
    // ---- 8 WMMAs, A fragment reused across the 128-wide stripe ----
    #pragma unroll
    for (int jt = 0; jt < 8; jt++) {
      Frag16 b;
      int bn = jt * 16 + l15;
      #pragma unroll
      for (int j = 0; j < 8; j++) {
        unsigned int pr = *(const unsigned int*)&Bsc[bn * LDS_S + ((j < 4) ? (bkb + 2 * j) : (16 + bkb + 2 * j - 8))];
        b.u[2 * j]     = (unsigned short)(pr & 0xffffu);
        b.u[2 * j + 1] = (unsigned short)(pr >> 16);
      }
      acc[jt] = __builtin_amdgcn_wmma_f32_16x16x32_bf16(
          false, a.v, false, b.v, (short)0, acc[jt], false, false);
    }

    if (wave == 0 && s + 1 < nsteps) __builtin_amdgcn_s_wait_tensorcnt((short)0);
    __syncthreads();
  }

  // ---- epilogue: C layout per ISA (VGPR r <-> M = r + 8*lanehalf) ----
  const int crow0 = m0 + wave * 16 + lhalf * 8;
  #pragma unroll
  for (int jt = 0; jt < 8; jt++) {
    int col = n0 + jt * 16 + l15;
    float bv = bias ? bias[col] : 0.0f;
    #pragma unroll
    for (int r = 0; r < 8; r++) {
      float v = acc[jt][r] + bv;
      if (relu) v = v > 0.0f ? v : 0.0f;
      if (bf16out) ((unsigned short*)C)[(size_t)(crow0 + r) * ldc + col] = f2bf(v);
      else         ((float*)C)[(size_t)(crow0 + r) * ldc + col] = v;
    }
  }
}

// =====================================================================
// f32 -> bf16 pack (vectorized, 4 elems/thread)
// =====================================================================
__global__ __launch_bounds__(256)
void hybridgcn_cvt(const float* __restrict__ src, unsigned short* __restrict__ dst)
{
  int idx = blockIdx.x * 256 + threadIdx.x;
  float4 v = ((const float4*)src)[idx];
  ushort4 o;
  o.x = f2bf(v.x); o.y = f2bf(v.y); o.z = f2bf(v.z); o.w = f2bf(v.w);
  ((ushort4*)dst)[idx] = o;
}

// =====================================================================
// Weight transpose + f32 -> bf16:  WT[n][k] = bf16(W[k][n]),  W is [Kd][Nd].
// k fastest in idx => coalesced writes; reads are L2-resident.
// =====================================================================
__global__ __launch_bounds__(256)
void hybridgcn_cvtT(const float* __restrict__ W, unsigned short* __restrict__ WT,
                    int Kd, int Nd)
{
  int idx = blockIdx.x * 256 + threadIdx.x;
  int n = idx / Kd, k = idx - n * Kd;
  WT[idx] = f2bf(W[(size_t)k * Nd + n]);
}

// =====================================================================
// Init: gat = broadcast(b_gat); emax = -inf; denom = 0.  Grid = NM/256.
// =====================================================================
__global__ __launch_bounds__(256)
void hybridgcn_init(float* __restrict__ gat, const float* __restrict__ b_gat,
                    float* __restrict__ emax, float* __restrict__ den)
{
  int idx = blockIdx.x * 256 + threadIdx.x;
  gat[idx] = b_gat[idx & (DINC - 1)];
  if (idx < NN * 4) { emax[idx] = -INFINITY; den[idx] = 0.0f; }
}

// =====================================================================
// GIN aggregation: h[src] += x[dst] per edge (h pre-set to x).
// =====================================================================
__global__ __launch_bounds__(256)
void hybridgcn_agg(const float* __restrict__ x, const int* __restrict__ ei,
                   float* __restrict__ h)
{
  int idx = blockIdx.x * 256 + threadIdx.x;
  int c = idx & 63, e = idx >> 6;
  int src = ei[e], dst = ei[EE + e];
  const float4 v = *(const float4*)(x + (size_t)dst * DINC + c * 4);
  float* p = h + (size_t)src * DINC + c * 4;
  atomicAdd(p + 0, v.x); atomicAdd(p + 1, v.y);
  atomicAdd(p + 2, v.z); atomicAdd(p + 3, v.w);
}

// =====================================================================
// GATv2 pass 1: per-(edge, head) score; atomic max into emax[dst][h].
// =====================================================================
__global__ __launch_bounds__(256)
void hybridgcn_scores(const float* __restrict__ xl, const float* __restrict__ xr,
                      const float* __restrict__ att, const int* __restrict__ ei,
                      float* __restrict__ esc, float* __restrict__ emax)
{
  int idx = blockIdx.x * 256 + threadIdx.x;
  int hh = idx & 3, e = idx >> 2;
  int src, dst;
  if (e < EE) { src = ei[e]; dst = ei[EE + e]; } else { src = dst = e - EE; }
  const float4* pl = (const float4*)(xl + (size_t)src * DINC + hh * 64);
  const float4* pr = (const float4*)(xr + (size_t)dst * DINC + hh * 64);
  const float4* pa = (const float4*)(att + hh * 64);
  float s = 0.0f;
  #pragma unroll
  for (int i = 0; i < 16; i++) {
    float4 a = pl[i], b = pr[i], w = pa[i];
    float m0 = a.x + b.x, m1 = a.y + b.y, m2 = a.z + b.z, m3 = a.w + b.w;
    m0 = m0 > 0.0f ? m0 : 0.2f * m0;
    m1 = m1 > 0.0f ? m1 : 0.2f * m1;
    m2 = m2 > 0.0f ? m2 : 0.2f * m2;
    m3 = m3 > 0.0f ? m3 : 0.2f * m3;
    s += m0 * w.x + m1 * w.y + m2 * w.z + m3 * w.w;
  }
  esc[idx] = s;
  atomicMaxF(&emax[dst * 4 + hh], s);
}

// =====================================================================
// GATv2 pass 2: ex = exp(e - emax[dst]); denom[dst] += ex. (in-place esc)
// =====================================================================
__global__ __launch_bounds__(256)
void hybridgcn_exp(const int* __restrict__ ei, float* __restrict__ esc,
                   const float* __restrict__ emax, float* __restrict__ den)
{
  int idx = blockIdx.x * 256 + threadIdx.x;
  int hh = idx & 3, e = idx >> 2;
  int dst = (e < EE) ? ei[EE + e] : (e - EE);
  float v = __expf(esc[idx] - emax[dst * 4 + hh]);
  esc[idx] = v;
  atomicAdd(&den[dst * 4 + hh], v);
}

// =====================================================================
// GATv2 pass 3: gat[dst] += (ex/denom[dst]) * xl[src].
// =====================================================================
__global__ __launch_bounds__(256)
void hybridgcn_scatter(const float* __restrict__ xl, const int* __restrict__ ei,
                       const float* __restrict__ esc, const float* __restrict__ den,
                       float* __restrict__ gat)
{
  int idx = blockIdx.x * 256 + threadIdx.x;
  int c = idx & 15, hh = (idx >> 4) & 3, e = idx >> 6;
  int src, dst;
  if (e < EE) { src = ei[e]; dst = ei[EE + e]; } else { src = dst = e - EE; }
  float alpha = esc[(e << 2) + hh] / den[dst * 4 + hh];
  const float4 v = *(const float4*)(xl + (size_t)src * DINC + hh * 64 + c * 4);
  float* g = gat + (size_t)dst * DINC + hh * 64 + c * 4;
  atomicAdd(g + 0, alpha * v.x); atomicAdd(g + 1, alpha * v.y);
  atomicAdd(g + 2, alpha * v.z); atomicAdd(g + 3, alpha * v.w);
}

// =====================================================================
// In-place LayerNorm over rows of 512. One block per row.
// =====================================================================
__global__ __launch_bounds__(256)
void hybridgcn_ln(float* __restrict__ out, const float* __restrict__ g,
                  const float* __restrict__ b)
{
  __shared__ float s1[256], s2[256];
  int row = blockIdx.x, t = threadIdx.x;
  float* p = out + (size_t)row * HIDC;
  float a0 = p[t], a1 = p[t + 256];
  s1[t] = a0 + a1;
  s2[t] = a0 * a0 + a1 * a1;
  __syncthreads();
  for (int off = 128; off > 0; off >>= 1) {
    if (t < off) { s1[t] += s1[t + off]; s2[t] += s2[t + off]; }
    __syncthreads();
  }
  float mean = s1[0] * (1.0f / HIDC);
  float var  = s2[0] * (1.0f / HIDC) - mean * mean;
  float inv  = rsqrtf(var + 1e-5f);
  p[t]       = (a0 - mean) * inv * g[t]       + b[t];
  p[t + 256] = (a1 - mean) * inv * g[t + 256] + b[t + 256];
}

// =====================================================================
extern "C" void kernel_launch(void* const* d_in, const int* in_sizes, int n_in,
                              void* d_out, int out_size, void* d_ws, size_t ws_size,
                              hipStream_t stream)
{
  (void)in_sizes; (void)n_in; (void)out_size; (void)ws_size;
  const float* x     = (const float*)d_in[0];
  const float* W_l   = (const float*)d_in[1];
  const float* W_r   = (const float*)d_in[2];
  const float* att   = (const float*)d_in[3];
  const float* b_gat = (const float*)d_in[4];
  const float* gW1   = (const float*)d_in[5];
  const float* gb1   = (const float*)d_in[6];
  const float* gW2   = (const float*)d_in[7];
  const float* gb2   = (const float*)d_in[8];
  const float* fW    = (const float*)d_in[9];
  const float* fb    = (const float*)d_in[10];
  const float* lng   = (const float*)d_in[11];
  const float* lnb   = (const float*)d_in[12];
  const int*   ei    = (const int*)d_in[13];
  float* out = (float*)d_out;

  // ---- workspace layout: f32 region then bf16 region (~64 MB total) ----
  float* ws   = (float*)d_ws;
  float* xl   = ws;                       // 8192x256 f32
  float* xr   = xl  + NM;
  float* h    = xr  + NM;                 // x + adj@x (f32, atomics)
  float* gat  = h   + NM;                 // GAT accum (f32, atomics)
  float* esc  = gat + NM;                 // (E+N)*4
  float* emax = esc + (size_t)ELOOP * 4;  // 8192*4
  float* den  = emax + NN * 4;            // 8192*4
  unsigned short* us    = (unsigned short*)(den + NN * 4);
  unsigned short* xbf   = us;             // bf16 copies of GEMM A operands
  unsigned short* hbf   = xbf   + NM;
  unsigned short* ttbf  = hbf   + NM;     // GIN hidden (bf16, GEMM-produced)
  unsigned short* ginbf = ttbf  + NM;
  unsigned short* gatbf = ginbf + NM;
  unsigned short* WlT   = gatbf + NM;     // pre-transposed bf16 weights [N][K]
  unsigned short* WrT   = WlT + DINC * DINC;
  unsigned short* g1T   = WrT + DINC * DINC;
  unsigned short* g2T   = g1T + DINC * DINC;
  unsigned short* fWT   = g2T + DINC * DINC;   // 512x512

  // ---- operand preparation (bf16 packs + weight transposes) ----
  hybridgcn_cvt <<<NM / 1024, 256, 0, stream>>>(x, xbf);
  hybridgcn_cvtT<<<(DINC * DINC) / 256, 256, 0, stream>>>(W_l, WlT, DINC, DINC);
  hybridgcn_cvtT<<<(DINC * DINC) / 256, 256, 0, stream>>>(W_r, WrT, DINC, DINC);
  hybridgcn_cvtT<<<(DINC * DINC) / 256, 256, 0, stream>>>(gW1, g1T, DINC, DINC);
  hybridgcn_cvtT<<<(DINC * DINC) / 256, 256, 0, stream>>>(gW2, g2T, DINC, DINC);
  hybridgcn_cvtT<<<(HIDC * HIDC) / 256, 256, 0, stream>>>(fW,  fWT, HIDC, HIDC);

  // h = x (then scatter-add adj@x); init gat/emax/den
  hipMemcpyAsync(h, x, (size_t)NM * sizeof(float), hipMemcpyDeviceToDevice, stream);
  hybridgcn_init<<<NM / 256, 256, 0, stream>>>(gat, b_gat, emax, den);

  const dim3 blk(256);
  // xl = x @ W_l, xr = x @ W_r  (TDM-staged bf16 WMMA)
  hybridgcn_gemm<<<dim3(DINC / 128, NN / 128), blk, 0, stream>>>(
      xbf, DINC, xbf, DINC, DINC, WlT, nullptr, xl, DINC, DINC, 0, 0);
  hybridgcn_gemm<<<dim3(DINC / 128, NN / 128), blk, 0, stream>>>(
      xbf, DINC, xbf, DINC, DINC, WrT, nullptr, xr, DINC, DINC, 0, 0);

  // GIN: h += adj@x ; tt = relu(h@W1+b1) [bf16 out] ; gin = tt@W2+b2 [bf16 out]
  hybridgcn_agg<<<(EE * 64) / 256, 256, 0, stream>>>(x, ei, h);
  hybridgcn_cvt<<<NM / 1024, 256, 0, stream>>>(h, hbf);
  hybridgcn_gemm<<<dim3(DINC / 128, NN / 128), blk, 0, stream>>>(
      hbf, DINC, hbf, DINC, DINC, g1T, gb1, ttbf, DINC, DINC, 1, 1);
  hybridgcn_gemm<<<dim3(DINC / 128, NN / 128), blk, 0, stream>>>(
      ttbf, DINC, ttbf, DINC, DINC, g2T, gb2, ginbf, DINC, DINC, 0, 1);

  // GATv2 softmax + scatter, then bf16 pack of gat
  hybridgcn_scores<<<(ELOOP * 4) / 256, 256, 0, stream>>>(xl, xr, att, ei, esc, emax);
  hybridgcn_exp<<<(ELOOP * 4) / 256, 256, 0, stream>>>(ei, esc, emax, den);
  hybridgcn_scatter<<<(ELOOP * 64) / 256, 256, 0, stream>>>(xl, ei, esc, den, gat);
  hybridgcn_cvt<<<NM / 1024, 256, 0, stream>>>(gat, gatbf);

  // fusion: out = relu([gat|gin] @ fus_W + fus_b) via split-A, then LN in place
  hybridgcn_gemm<<<dim3(HIDC / 128, NN / 128), blk, 0, stream>>>(
      gatbf, DINC, ginbf, DINC, DINC, fWT, fb, out, HIDC, HIDC, 1, 0);
  hybridgcn_ln<<<NN, 256, 0, stream>>>(out, lng, lnb);
}
